// VariableSelectionNetwork_70196945485886
// MI455X (gfx1250) — matrix operand order repaired and checked by reading, not compiled, not run on hardware
//
#include <hip/hip_runtime.h>
#include <hip/hip_bf16.h>

typedef __attribute__((ext_vector_type(16))) _Float16 v16h;
typedef __attribute__((ext_vector_type(8)))  float    v8f;

#define BT   32768
#define FDIM 32
#define UDIM 64
#define LNEPS 1e-3f

__device__ __forceinline__ float eluf(float x)  { return x > 0.f ? x : __expf(x) - 1.f; }
__device__ __forceinline__ float sigf(float x)  { return 1.f / (1.f + __expf(-x)); }

__device__ __forceinline__ v8f wmma16(v16h a, v16h b, v8f c) {
  // D(f32 16x16) = A(f16 16x32) * B(f16 32x16) + C
  return __builtin_amdgcn_wmma_f32_16x16x32_f16(false, a, false, b, (short)0, c, false, false);
}

// Logical-K permutation: A slot (lane, e) holds logical column u = hi*16 + e
// (physical k = e + 8*(e>=8) + 8*hi). B slot (lane, e) holds the same logical
// row u, so A and B agree; brow() is not used by any load.
__device__ __forceinline__ int brow(int lane) {
  return ((lane >> 3) & 1) * 16 + (lane >> 4) * 8 + (lane & 7);
}

__device__ __forceinline__ float redsum16(float v) {
  v += __shfl_xor(v, 1);  v += __shfl_xor(v, 2);
  v += __shfl_xor(v, 4);  v += __shfl_xor(v, 8);
  return v;
}
__device__ __forceinline__ float redmax16(float v) {
  v = fmaxf(v, __shfl_xor(v, 1));  v = fmaxf(v, __shfl_xor(v, 2));
  v = fmaxf(v, __shfl_xor(v, 4));  v = fmaxf(v, __shfl_xor(v, 8));
  return v;
}

// ---------------------------------------------------------------------------
// Stage A: variable-weights GRN + softmax  ->  w [BT, 32]
// One wave per 16-row tile; 8 waves per block; grid = BT/16/8 = 256 blocks.
// ---------------------------------------------------------------------------
__global__ __launch_bounds__(256) void vsn_weights_kernel(
    const float* __restrict__ x,
    const float* __restrict__ w1w, const float* __restrict__ b1w,
    const float* __restrict__ w2w, const float* __restrict__ b2w,
    const float* __restrict__ wg1w, const float* __restrict__ bg1w,
    const float* __restrict__ wg2w, const float* __restrict__ bg2w,
    const float* __restrict__ gammaw, const float* __restrict__ betaw,
    float* __restrict__ wout)
{
  __shared__ __align__(32) _Float16 sW1[FDIM * FDIM], sW2[FDIM * FDIM];
  __shared__ __align__(32) _Float16 sG1[FDIM * FDIM], sG2[FDIM * FDIM];
  __shared__ float sb1[FDIM], sb2[FDIM], sbg1[FDIM], sbg2[FDIM], sgam[FDIM], sbet[FDIM];
  __shared__ float sX[8][16 * FDIM];                    // per-wave x tile (f32)
  __shared__ __align__(32) _Float16 sH[8][16 * FDIM];   // per-wave D->A staging

  const int tid = threadIdx.x;
  for (int i = tid; i < FDIM * FDIM; i += 256) {
    sW1[i] = (_Float16)w1w[(i % FDIM) * FDIM + i / FDIM];
    sW2[i] = (_Float16)w2w[(i % FDIM) * FDIM + i / FDIM];
    sG1[i] = (_Float16)wg1w[(i % FDIM) * FDIM + i / FDIM];
    sG2[i] = (_Float16)wg2w[(i % FDIM) * FDIM + i / FDIM];
  }
  if (tid < FDIM) {
    sb1[tid]  = b1w[tid];  sb2[tid]  = b2w[tid];
    sbg1[tid] = bg1w[tid]; sbg2[tid] = bg2w[tid];
    sgam[tid] = gammaw[tid]; sbet[tid] = betaw[tid];
  }

  const int wave = tid >> 5, lane = tid & 31;
  const int mi = lane & 15, hi = lane >> 4;
  const int row0 = (blockIdx.x * 8 + wave) * 16;
  const int br = brow(lane);  // unused: B loads use k = hi*16 + e

  // x tile -> A regs (logical u = hi*16 + e : 16 consecutive columns) + f32 LDS copy
  v16h xa;
#pragma unroll
  for (int e = 0; e < 16; ++e) {
    const int u = hi * 16 + e;
    const float v = x[(row0 + mi) * FDIM + u];
    xa[e] = (_Float16)v;
    sX[wave][mi * FDIM + u] = v;
  }
  __syncthreads();

  // h = elu(x @ w1w + b1w)
  v8f hD[2];
#pragma unroll
  for (int t = 0; t < 2; ++t) {
    v8f c; const float cb = sb1[t * 16 + mi];
#pragma unroll
    for (int r = 0; r < 8; ++r) c[r] = cb;
    const v16h Bm = *(const v16h*)&sW1[(t * 16 + mi) * FDIM + hi * 16];
    hD[t] = wmma16(xa, Bm, c);
  }
#pragma unroll
  for (int t = 0; t < 2; ++t)
#pragma unroll
    for (int r = 0; r < 8; ++r)
      sH[wave][(r + 8 * hi) * FDIM + t * 16 + mi] = (_Float16)eluf(hD[t][r]);
  __syncthreads();

  const v16h ha = *(const v16h*)&sH[wave][mi * FDIM + hi * 16];

  // h2 = h @ w2w + b2w
  v8f h2D[2];
#pragma unroll
  for (int t = 0; t < 2; ++t) {
    v8f c; const float cb = sb2[t * 16 + mi];
#pragma unroll
    for (int r = 0; r < 8; ++r) c[r] = cb;
    const v16h Bm = *(const v16h*)&sW2[(t * 16 + mi) * FDIM + hi * 16];
    h2D[t] = wmma16(ha, Bm, c);
  }
  __syncthreads();
#pragma unroll
  for (int t = 0; t < 2; ++t)
#pragma unroll
    for (int r = 0; r < 8; ++r)
      sH[wave][(r + 8 * hi) * FDIM + t * 16 + mi] = (_Float16)h2D[t][r];
  __syncthreads();

  const v16h h2a = *(const v16h*)&sH[wave][mi * FDIM + hi * 16];

  // g = (h2 @ wg1w + bg1w) * sigmoid(h2 @ wg2w + bg2w)
  v8f g1D[2], g2D[2];
#pragma unroll
  for (int t = 0; t < 2; ++t) {
    v8f c1, c2;
    const float cb1 = sbg1[t * 16 + mi], cb2 = sbg2[t * 16 + mi];
#pragma unroll
    for (int r = 0; r < 8; ++r) { c1[r] = cb1; c2[r] = cb2; }
    const v16h B1 = *(const v16h*)&sG1[(t * 16 + mi) * FDIM + hi * 16];
    const v16h B2 = *(const v16h*)&sG2[(t * 16 + mi) * FDIM + hi * 16];
    g1D[t] = wmma16(h2a, B1, c1);
    g2D[t] = wmma16(h2a, B2, c2);
  }

  // s = g + x ; LayerNorm over F=32 ; softmax over F=32
  float sv[2][8];
#pragma unroll
  for (int t = 0; t < 2; ++t)
#pragma unroll
    for (int r = 0; r < 8; ++r) {
      const int m = r + 8 * hi, n = t * 16 + mi;
      sv[t][r] = g1D[t][r] * sigf(g2D[t][r]) + sX[wave][m * FDIM + n];
    }

#pragma unroll
  for (int r = 0; r < 8; ++r) {
    const float sm = redsum16(sv[0][r] + sv[1][r]);
    const float sq = redsum16(sv[0][r] * sv[0][r] + sv[1][r] * sv[1][r]);
    const float mean = sm * (1.f / 32.f);
    const float var  = sq * (1.f / 32.f) - mean * mean;
    const float inv  = rsqrtf(var + LNEPS);
    const int n0 = mi, n1 = 16 + mi;
    const float y0 = (sv[0][r] - mean) * inv * sgam[n0] + sbet[n0];
    const float y1 = (sv[1][r] - mean) * inv * sgam[n1] + sbet[n1];
    const float mx = redmax16(fmaxf(y0, y1));
    const float e0 = __expf(y0 - mx), e1 = __expf(y1 - mx);
    const float is = 1.f / redsum16(e0 + e1);
    const int m = r + 8 * hi;
    wout[(row0 + m) * FDIM + n0] = e0 * is;
    wout[(row0 + m) * FDIM + n1] = e1 * is;
  }
}

// ---------------------------------------------------------------------------
// Stage B: per-feature GRNs + weighted sum  ->  out [BT, 64]
// Block = 8 waves; each wave owns one 16-row tile, loops over all 32 features;
// per-feature 64x64 weights staged once per block in LDS as f16.
// ---------------------------------------------------------------------------
__global__ __launch_bounds__(256) void vsn_main_kernel(
    const float* __restrict__ x,
    const float* __restrict__ W1,  const float* __restrict__ b1,
    const float* __restrict__ W2,  const float* __restrict__ b2,
    const float* __restrict__ Wg1, const float* __restrict__ bg1,
    const float* __restrict__ Wg2, const float* __restrict__ bg2,
    const float* __restrict__ Wp,  const float* __restrict__ bp,
    const float* __restrict__ gamma, const float* __restrict__ beta,
    const float* __restrict__ wsel,  // [BT,32] softmax weights from stage A
    float* __restrict__ out)
{
  __shared__ __align__(32) _Float16 sW2[UDIM * UDIM], sG1[UDIM * UDIM], sG2[UDIM * UDIM];
  __shared__ float sV[9][UDIM];            // W1,b1,b2,bg1,bg2,Wp,bp,gamma,beta
  __shared__ float sXv[8][16], sWt[8][16]; // per-wave scalars x[:,f], w[:,f]
  __shared__ __align__(32) _Float16 sH2[8][16 * UDIM];   // per-wave D->A staging

  const int tid = threadIdx.x;
  const int wave = tid >> 5, lane = tid & 31;
  const int mi = lane & 15, hi = lane >> 4;
  const int row0 = (blockIdx.x * 8 + wave) * 16;
  const int br = brow(lane);

  v8f acc[4];
#pragma unroll
  for (int t = 0; t < 4; ++t)
#pragma unroll
    for (int r = 0; r < 8; ++r) acc[t][r] = 0.f;

  for (int f = 0; f < FDIM; ++f) {
    __syncthreads();  // protect LDS reuse (WAR from previous iteration)
    {
      const float* W2f = W2  + (size_t)f * UDIM * UDIM;
      const float* G1f = Wg1 + (size_t)f * UDIM * UDIM;
      const float* G2f = Wg2 + (size_t)f * UDIM * UDIM;
      for (int i = tid; i < UDIM * UDIM; i += 256) {
        sW2[i] = (_Float16)W2f[(i % UDIM) * UDIM + i / UDIM];
        sG1[i] = (_Float16)G1f[(i % UDIM) * UDIM + i / UDIM];
        sG2[i] = (_Float16)G2f[(i % UDIM) * UDIM + i / UDIM];
      }
      if (tid < UDIM) {
        sV[0][tid] = W1 [f * UDIM + tid];
        sV[1][tid] = b1 [f * UDIM + tid];
        sV[2][tid] = b2 [f * UDIM + tid];
        sV[3][tid] = bg1[f * UDIM + tid];
        sV[4][tid] = bg2[f * UDIM + tid];
        sV[5][tid] = Wp [f * UDIM + tid];
        sV[6][tid] = bp [f * UDIM + tid];
        sV[7][tid] = gamma[f * UDIM + tid];
        sV[8][tid] = beta [f * UDIM + tid];
      }
      if (lane < 16) {
        sXv[wave][lane] = x   [(row0 + lane) * FDIM + f];
        sWt[wave][lane] = wsel[(row0 + lane) * FDIM + f];
      }
    }
    __syncthreads();

    const float xv = sXv[wave][mi];

    // h = elu(x_f * W1 + b1), built directly in A layout.
    // Logical column for slot (lane, e, chunk c) is u = c*32 + hi*16 + e.
    v16h ha[2];
#pragma unroll
    for (int c = 0; c < 2; ++c)
#pragma unroll
      for (int e = 0; e < 16; ++e) {
        const int u = c * 32 + hi * 16 + e;
        ha[c][e] = (_Float16)eluf(xv * sV[0][u] + sV[1][u]);
      }

    // h2 = h @ W2 + b2  (4 N-tiles x 2 K-chunks = 8 WMMA)
    v8f h2D[4];
#pragma unroll
    for (int t = 0; t < 4; ++t) {
      v8f cc; const float cb = sV[2][t * 16 + mi];
#pragma unroll
      for (int r = 0; r < 8; ++r) cc[r] = cb;
#pragma unroll
      for (int c = 0; c < 2; ++c) {
        const v16h Bm = *(const v16h*)&sW2[(t * 16 + mi) * UDIM + c * 32 + hi * 16];
        cc = wmma16(ha[c], Bm, cc);
      }
      h2D[t] = cc;
    }

    // D-layout -> A-layout via per-wave LDS scratch (plain [m][v] row-major;
    // readback is contiguous thanks to the logical-K permutation)
#pragma unroll
    for (int t = 0; t < 4; ++t)
#pragma unroll
      for (int r = 0; r < 8; ++r)
        sH2[wave][(r + 8 * hi) * UDIM + t * 16 + mi] = (_Float16)h2D[t][r];
    __syncthreads();

    v16h h2a[2];
#pragma unroll
    for (int c = 0; c < 2; ++c)
      h2a[c] = *(const v16h*)&sH2[wave][mi * UDIM + c * 32 + hi * 16];

    // g1 = h2 @ Wg1 + bg1 ; g2 = h2 @ Wg2 + bg2  (16 WMMA)
    v8f g1D[4], g2D[4];
#pragma unroll
    for (int t = 0; t < 4; ++t) {
      v8f c1, c2;
      const float cb1 = sV[3][t * 16 + mi], cb2 = sV[4][t * 16 + mi];
#pragma unroll
      for (int r = 0; r < 8; ++r) { c1[r] = cb1; c2[r] = cb2; }
#pragma unroll
      for (int c = 0; c < 2; ++c) {
        const v16h B1 = *(const v16h*)&sG1[(t * 16 + mi) * UDIM + c * 32 + hi * 16];
        const v16h B2 = *(const v16h*)&sG2[(t * 16 + mi) * UDIM + c * 32 + hi * 16];
        c1 = wmma16(h2a[c], B1, c1);
        c2 = wmma16(h2a[c], B2, c2);
      }
      g1D[t] = c1; g2D[t] = c2;
    }

    float xvm[8], wtm[8];
#pragma unroll
    for (int r = 0; r < 8; ++r) {
      const int m = r + 8 * hi;
      xvm[r] = sXv[wave][m];
      wtm[r] = sWt[wave][m];
    }

    // s = GLU(g1,g2) + (x_f * Wp + bp) ; LayerNorm over U=64 ; weighted accumulate
    float sv[4][8];
#pragma unroll
    for (int t = 0; t < 4; ++t)
#pragma unroll
      for (int r = 0; r < 8; ++r) {
        const int v = t * 16 + mi;
        sv[t][r] = g1D[t][r] * sigf(g2D[t][r]) + xvm[r] * sV[5][v] + sV[6][v];
      }

#pragma unroll
    for (int r = 0; r < 8; ++r) {
      float sm = 0.f, sq = 0.f;
#pragma unroll
      for (int t = 0; t < 4; ++t) { sm += sv[t][r]; sq += sv[t][r] * sv[t][r]; }
      sm = redsum16(sm); sq = redsum16(sq);
      const float mean = sm * (1.f / 64.f);
      const float var  = sq * (1.f / 64.f) - mean * mean;
      const float inv  = rsqrtf(var + LNEPS);
      const float wr   = wtm[r];
#pragma unroll
      for (int t = 0; t < 4; ++t) {
        const int v = t * 16 + mi;
        const float y = (sv[t][r] - mean) * inv * sV[7][v] + sV[8][v];
        acc[t][r] += y * wr;
      }
    }
  }

  // out[row, u]
#pragma unroll
  for (int t = 0; t < 4; ++t)
#pragma unroll
    for (int r = 0; r < 8; ++r)
      out[(size_t)(row0 + r + 8 * hi) * UDIM + t * 16 + mi] = acc[t][r];
}

extern "C" void kernel_launch(void* const* d_in, const int* in_sizes, int n_in,
                              void* d_out, int out_size, void* d_ws, size_t ws_size,
                              hipStream_t stream) {
  (void)in_sizes; (void)n_in; (void)out_size; (void)d_ws; (void)ws_size;

  const float* x     = (const float*)d_in[0];
  const float* W1    = (const float*)d_in[1];
  const float* b1    = (const float*)d_in[2];
  const float* W2    = (const float*)d_in[3];
  const float* b2    = (const float*)d_in[4];
  const float* Wg1   = (const float*)d_in[5];
  const float* bg1   = (const float*)d_in[6];
  const float* Wg2   = (const float*)d_in[7];
  const float* bg2   = (const float*)d_in[8];
  const float* Wp    = (const float*)d_in[9];
  const float* bp    = (const float*)d_in[10];
  const float* gamma = (const float*)d_in[11];
  const float* beta  = (const float*)d_in[12];
  const float* w1w   = (const float*)d_in[13];
  const float* b1w   = (const float*)d_in[14];
  const float* w2w   = (const float*)d_in[15];
  const float* b2w   = (const float*)d_in[16];
  const float* wg1w  = (const float*)d_in[17];
  const float* bg1w  = (const float*)d_in[18];
  const float* wg2w  = (const float*)d_in[19];
  const float* bg2w  = (const float*)d_in[20];
  const float* gmw   = (const float*)d_in[21];
  const float* btw   = (const float*)d_in[22];

  float* out  = (float*)d_out;                 // [BT, 64]
  float* wout = out + (size_t)BT * UDIM;       // [BT, 32]

  const int tiles  = BT / 16;                  // 2048
  const int blocks = tiles / 8;                // 256 blocks of 8 waves

  vsn_weights_kernel<<<blocks, 256, 0, stream>>>(
      x, w1w, b1w, w2w, b2w, wg1w, bg1w, wg2w, bg2w, gmw, btw, wout);

  vsn_main_kernel<<<blocks, 256, 0, stream>>>(
      x, W1, b1, W2, b2, Wg1, bg1, Wg2, bg2, Wp, bp, gamma, beta, wout, out);
}
